// WaveGlow_53747220742688
// MI455X (gfx1250) — matrix operand order repaired and checked by reading, not compiled, not run on hardware
//
#include <hip/hip_runtime.h>
#include <cstddef>
#include <cstdint>

typedef __attribute__((ext_vector_type(16))) _Float16 v16h;
typedef __attribute__((ext_vector_type(8)))  _Float16 v8h;
typedef __attribute__((ext_vector_type(8)))  float    v8f;

#define TG    2048
#define BATCH 4
#define NCH   256
#define NMEL  80
#define CONDK 640
#define NKK_LAYER 44   /* K=1408 = 44*32 */
#define NKK_RS    8    /* K=256  = 8*32  */
#define HPAD  128      /* >= max dilation (2^7) */
#define HPITCH (TG + 2 * HPAD)   /* 2304, padded f16 h row */

__device__ __forceinline__ v8f wmma16(v16h a, v16h b, v8f c) {
  return __builtin_amdgcn_wmma_f32_16x16x32_f16(false, a, false, b, (short)0, c,
                                                false, false);
}

// ---------------------------------------------------------------------------
// elementwise / setup kernels
// ---------------------------------------------------------------------------
__global__ __launch_bounds__(256) void zero_kernel(float* p, int n) {
  int i = blockIdx.x * blockDim.x + threadIdx.x;
  if (i < n) p[i] = 0.0f;
}
__global__ __launch_bounds__(256) void zero_h16_kernel(_Float16* p, int n) {
  int i = blockIdx.x * blockDim.x + threadIdx.x;
  if (i < n) p[i] = (_Float16)0.0f;
}

// transposed conv1d upsampler -> grouped spect, stored f16 for WMMA B operand
__global__ __launch_bounds__(256) void upsample_kernel(const float* __restrict__ x,
                                                       const float* __restrict__ up_w,
                                                       const float* __restrict__ up_b,
                                                       _Float16* __restrict__ spect16) {
  int tid = blockIdx.x * blockDim.x + threadIdx.x;
  if (tid >= BATCH * NMEL * 16384) return;
  int t = tid & 16383;
  int m = (tid >> 14) % NMEL;
  int b = tid / (NMEL * 16384);
  int s1 = t >> 8;           // HOP = 256
  int s0 = s1 - 3;           // WIN/HOP = 4 taps
  if (s0 < 0) s0 = 0;
  if (s1 > 63) s1 = 63;
  float acc = up_b[m];
  for (int s = s0; s <= s1; ++s) {
    int kidx = t - (s << 8);
    for (int i = 0; i < NMEL; ++i)
      acc += x[((size_t)b * NMEL + i) * 64 + s] *
             up_w[((size_t)i * NMEL + m) * 1024 + kidx];
  }
  int g = t & 7, tg = t >> 3;
  spect16[((size_t)b * (NMEL * 8) + m * 8 + g) * TG + tg] = (_Float16)acc;
}

__global__ __launch_bounds__(256) void audio_init_kernel(const float* __restrict__ y,
                                                         float* __restrict__ audio) {
  int tid = blockIdx.x * blockDim.x + threadIdx.x;
  if (tid >= BATCH * 8 * TG) return;
  int t = tid & (TG - 1);
  int g = (tid >> 11) & 7;
  int b = tid >> 14;
  audio[((size_t)b * 8 + g) * TG + t] = y[(size_t)b * 16384 + t * 8 + g];
}

__global__ __launch_bounds__(256) void early_split_kernel(const float* __restrict__ in,
                                                          float* __restrict__ dst,
                                                          float* __restrict__ zout,
                                                          int zbase, int n_old) {
  int tid = blockIdx.x * blockDim.x + threadIdx.x;
  if (tid >= BATCH * n_old * TG) return;
  int t = tid & (TG - 1);
  int ch = (tid >> 11) % n_old;
  int b = tid / (n_old * TG);
  float v = in[((size_t)b * n_old + ch) * TG + t];
  if (ch < 2)
    zout[(size_t)b * 8 * TG + (zbase + ch) * TG + t] = v;
  else
    dst[((size_t)b * (n_old - 2) + (ch - 2)) * TG + t] = v;
}

__global__ __launch_bounds__(256) void final_copy_kernel(const float* __restrict__ in,
                                                         float* __restrict__ zout) {
  int tid = blockIdx.x * blockDim.x + threadIdx.x;
  if (tid >= BATCH * 4 * TG) return;
  int t = tid & (TG - 1);
  int ch = (tid >> 11) & 3;
  int b = tid >> 13;
  zout[(size_t)b * 8 * TG + (4 + ch) * TG + t] = in[((size_t)b * 4 + ch) * TG + t];
}

__global__ __launch_bounds__(256) void wmult_kernel(const float* __restrict__ in,
                                                    float* __restrict__ out,
                                                    const float* __restrict__ W, int n) {
  int tid = blockIdx.x * blockDim.x + threadIdx.x;
  if (tid >= BATCH * TG) return;
  int t = tid & (TG - 1);
  int b = tid >> 11;
  float v[8];
  for (int c = 0; c < n; ++c) v[c] = in[((size_t)b * n + c) * TG + t];
  for (int o = 0; o < n; ++o) {
    float acc = 0.f;
    for (int c = 0; c < n; ++c) acc += W[o * n + c] * v[c];
    out[((size_t)b * n + o) * TG + t] = acc;
  }
}

// single-thread slogdet via partial-pivot Gaussian elimination (n <= 8)
__global__ void logdet_kernel(const float* __restrict__ W, float* acc, int n, float scale) {
  if (threadIdx.x != 0 || blockIdx.x != 0) return;
  float M[64];
  for (int i = 0; i < n * n; ++i) M[i] = W[i];
  float ls = 0.f;
  for (int c = 0; c < n; ++c) {
    int p = c; float best = fabsf(M[c * n + c]);
    for (int r = c + 1; r < n; ++r) {
      float a = fabsf(M[r * n + c]);
      if (a > best) { best = a; p = r; }
    }
    if (p != c)
      for (int j = 0; j < n; ++j) {
        float tv = M[c * n + j]; M[c * n + j] = M[p * n + j]; M[p * n + j] = tv;
      }
    float piv = M[c * n + c];
    ls += logf(fabsf(piv));
    for (int r = c + 1; r < n; ++r) {
      float f = M[r * n + c] / piv;
      for (int j = c; j < n; ++j) M[r * n + j] -= f * M[c * n + j];
    }
  }
  acc[1] += scale * ls;
}

// start conv 1x1: writes f32 master h and padded f16 copy
__global__ __launch_bounds__(256) void start_conv_kernel(const float* __restrict__ audio,
                                                         const float* __restrict__ sw,
                                                         const float* __restrict__ sb,
                                                         float* __restrict__ h32,
                                                         _Float16* __restrict__ h16,
                                                         int n_half, int n_rem) {
  int tid = blockIdx.x * blockDim.x + threadIdx.x;
  if (tid >= BATCH * NCH * TG) return;
  int t = tid & (TG - 1);
  int oc = (tid >> 11) & (NCH - 1);
  int b = tid >> 19;
  float acc = sb[oc];
  for (int ic = 0; ic < n_half; ++ic)
    acc += sw[oc * n_half + ic] * audio[((size_t)b * n_rem + ic) * TG + t];
  h32[((size_t)b * NCH + oc) * TG + t] = acc;
  h16[((size_t)b * NCH + oc) * HPITCH + HPAD + t] = (_Float16)acc;
}

// ---------------------------------------------------------------------------
// weight packing into WMMA A-fragment lane order (16x32 f16 tiles)
// lanes 0-15: K {0..15}; lanes 16-31: K {16..31}
// ---------------------------------------------------------------------------
__device__ __forceinline__ int afrag_klocal(int e, int lane) {
  int base = 0;
  if (lane & 16) base += 16;
  return base + e;
}

__global__ __launch_bounds__(256) void pack_layerA_kernel(const float* __restrict__ in_w,
                                                          const float* __restrict__ cond_w,
                                                          int layer,
                                                          _Float16* __restrict__ Apack) {
  int tid = blockIdx.x * blockDim.x + threadIdx.x;
  if (tid >= 32 * NKK_LAYER * 32 * 16) return;
  int e = tid & 15;
  int lane = (tid >> 4) & 31;
  int rest = tid >> 9;
  int kk = rest % NKK_LAYER;
  int mt = rest / NKK_LAYER;
  int K = kk * 32 + afrag_klocal(e, lane);
  int m = (mt << 4) + (lane & 15);
  float v;
  if (K < 768) {
    int tap = K >> 8;
    int ic = K & 255;
    v = in_w[((size_t)m * NCH + ic) * 3 + tap];
  } else {
    v = cond_w[((size_t)(layer * 512 + m)) * CONDK + (K - 768)];
  }
  Apack[tid] = (_Float16)v;
}

__global__ __launch_bounds__(256) void pack_rs_kernel(const float* __restrict__ rs_w,
                                                      _Float16* __restrict__ Rpack, int Mo) {
  int tid = blockIdx.x * blockDim.x + threadIdx.x;
  if (tid >= (Mo / 16) * NKK_RS * 32 * 16) return;
  int e = tid & 15;
  int lane = (tid >> 4) & 31;
  int rest = tid >> 9;
  int kk = rest & (NKK_RS - 1);
  int mt = rest >> 3;
  int K = kk * 32 + afrag_klocal(e, lane);
  int m = (mt << 4) + (lane & 15);
  Rpack[tid] = (_Float16)rs_w[(size_t)m * NCH + K];
}

__global__ void layer_bias_kernel(const float* __restrict__ in_b,
                                  const float* __restrict__ cond_b, int layer,
                                  float* __restrict__ bias) {
  int i = blockIdx.x * blockDim.x + threadIdx.x;
  if (i < 512) bias[i] = in_b[i] + cond_b[layer * 512 + i];
}

// ---------------------------------------------------------------------------
// WMMA GEMM #1: fused dilated-conv + conditioning.  M=512, K=1408, N=TG per batch.
// WG = 256 (8 waves); per stage: 64 K-rows x 32 N-cols staged in LDS, each wave
// computes one 16-row tile over both 16-col tiles -> 4 WMMAs per barrier pair.
// Source-select and tap are uniform per 64-row stage (boundaries 256/512/768).
// h16 is zero-padded by HPAD columns -> no bounds checks for dilated taps.
// ---------------------------------------------------------------------------
__global__ __launch_bounds__(256) void layer_gemm_kernel(const _Float16* __restrict__ Apack,
                                                         const _Float16* __restrict__ h16,
                                                         const _Float16* __restrict__ spect16,
                                                         const float* __restrict__ bias,
                                                         float* __restrict__ apre,
                                                         int dil) {
  __shared__ alignas(64) _Float16 ldsB[32][64];   // [n][k]
  int b = blockIdx.z;
  int t0 = blockIdx.y * 32;
  int wv = threadIdx.x >> 5, lane = threadIdx.x & 31;
  int mt = blockIdx.x * 8 + wv;                   // 16-row tile id (0..31)
  const _Float16* hb = h16 + (size_t)b * NCH * HPITCH;
  const _Float16* sb = spect16 + (size_t)b * CONDK * TG;
  int kr = threadIdx.x >> 2;                      // 0..63 (stage K row)
  int n0 = (threadIdx.x & 3) << 3;                // 0,8,16,24 (stage col base)
  int n = lane & 15;
  int ksel = (lane & 16) ? 16 : 0;
  v8f c0 = {0.f, 0.f, 0.f, 0.f, 0.f, 0.f, 0.f, 0.f};
  v8f c1 = c0;
  for (int s = 0; s < NKK_LAYER / 2; ++s) {
    // ---- stage 64x32 f16 tile ----
    int K = s * 64 + kr;
    const _Float16* src;
    if (K < 768) {                       // uniform per stage
      int tap = K >> 8;                  // uniform per stage
      int ic = K & 255;
      src = hb + (size_t)ic * HPITCH + (HPAD + t0 + n0 + (tap - 1) * dil);
    } else {
      src = sb + (size_t)(K - 768) * TG + (t0 + n0);
    }
#pragma unroll
    for (int j = 0; j < 8; ++j) ldsB[n0 + j][kr] = src[j];
    __syncthreads();
    // ---- compute ----
    size_t aoff = ((size_t)(mt * NKK_LAYER + 2 * s) * 32 + lane) * 16;
    v16h a0 = *(const v16h*)(Apack + aoff);
    v16h a1 = *(const v16h*)(Apack + aoff + 32 * 16);
    if (s + 1 < NKK_LAYER / 2)
      __builtin_prefetch((const void*)(Apack + aoff + 2 * 32 * 16), 0, 1);
    v16h b0a = *(const v16h*)&ldsB[n][ksel];
    v16h b0b = *(const v16h*)&ldsB[n][32 + ksel];
    v16h b1a = *(const v16h*)&ldsB[16 + n][ksel];
    v16h b1b = *(const v16h*)&ldsB[16 + n][32 + ksel];
    c0 = wmma16(a0, b0a, c0);
    c0 = wmma16(a1, b0b, c0);
    c1 = wmma16(a0, b1a, c1);
    c1 = wmma16(a1, b1b, c1);
    __syncthreads();
  }
  int mhalf = (lane & 16) ? 8 : 0;
  float* ob = apre + (size_t)b * 512 * TG;
#pragma unroll
  for (int r = 0; r < 8; ++r) {
    int m = mt * 16 + mhalf + r;
    float bm = bias[m];
    ob[(size_t)m * TG + t0 + n] = c0[r] + bm;
    ob[(size_t)m * TG + t0 + 16 + n] = c1[r] + bm;
  }
}

__global__ __launch_bounds__(256) void gate_kernel(const float* __restrict__ apre,
                                                   _Float16* __restrict__ acts16) {
  int tid = blockIdx.x * blockDim.x + threadIdx.x;
  if (tid >= BATCH * NCH * TG) return;
  int t = tid & (TG - 1);
  int ch = (tid >> 11) & (NCH - 1);
  int b = tid >> 19;
  const float* ab = apre + (size_t)b * 512 * TG;
  float ta = tanhf(ab[(size_t)ch * TG + t]);
  float sg = 1.0f / (1.0f + expf(-ab[(size_t)(NCH + ch) * TG + t]));
  acts16[((size_t)b * NCH + ch) * TG + t] = (_Float16)(ta * sg);
}

// WMMA GEMM #2: res/skip 1x1, K=256, fused residual epilogue (updates f32 h and
// its padded f16 shadow).  Staging is one aligned 16B v8h load per thread.
__global__ __launch_bounds__(256) void rs_gemm_kernel(const _Float16* __restrict__ Rpack,
                                                      const _Float16* __restrict__ acts16,
                                                      const float* __restrict__ rsb,
                                                      float* __restrict__ h32,
                                                      _Float16* __restrict__ h16,
                                                      float* __restrict__ wout,
                                                      int last) {
  __shared__ alignas(64) _Float16 ldsB[32][64];
  int b = blockIdx.z;
  int t0 = blockIdx.y * 32;
  int wv = threadIdx.x >> 5, lane = threadIdx.x & 31;
  int mt = blockIdx.x * 8 + wv;
  const _Float16* xb = acts16 + (size_t)b * NCH * TG;
  int kr = threadIdx.x >> 2;
  int n0 = (threadIdx.x & 3) << 3;
  int n = lane & 15;
  int ksel = (lane & 16) ? 16 : 0;
  v8f c0 = {0.f, 0.f, 0.f, 0.f, 0.f, 0.f, 0.f, 0.f};
  v8f c1 = c0;
  for (int s = 0; s < NKK_RS / 2; ++s) {
    v8h vals = *(const v8h*)(xb + (size_t)(s * 64 + kr) * TG + t0 + n0);
#pragma unroll
    for (int j = 0; j < 8; ++j) ldsB[n0 + j][kr] = vals[j];
    __syncthreads();
    size_t aoff = ((size_t)(mt * NKK_RS + 2 * s) * 32 + lane) * 16;
    v16h a0 = *(const v16h*)(Rpack + aoff);
    v16h a1 = *(const v16h*)(Rpack + aoff + 32 * 16);
    v16h b0a = *(const v16h*)&ldsB[n][ksel];
    v16h b0b = *(const v16h*)&ldsB[n][32 + ksel];
    v16h b1a = *(const v16h*)&ldsB[16 + n][ksel];
    v16h b1b = *(const v16h*)&ldsB[16 + n][32 + ksel];
    c0 = wmma16(a0, b0a, c0);
    c0 = wmma16(a1, b0b, c0);
    c1 = wmma16(a0, b1a, c1);
    c1 = wmma16(a1, b1b, c1);
    __syncthreads();
  }
  int mhalf = (lane & 16) ? 8 : 0;
  float* hb = h32 + (size_t)b * NCH * TG;
  _Float16* hs = h16 + (size_t)b * NCH * HPITCH;
  float* wb = wout + (size_t)b * NCH * TG;
#pragma unroll
  for (int r = 0; r < 8; ++r) {
    int m = mt * 16 + mhalf + r;
    float rb = rsb[m];
#pragma unroll
    for (int q = 0; q < 2; ++q) {
      size_t col = (size_t)t0 + q * 16 + n;
      float v = (q ? c1[r] : c0[r]) + rb;
      if (last) {
        wb[(size_t)m * TG + col] += v;
      } else if (m < NCH) {
        float nv = hb[(size_t)m * TG + col] + v;
        hb[(size_t)m * TG + col] = nv;
        hs[(size_t)m * HPITCH + HPAD + col] = (_Float16)nv;
      } else {
        wb[(size_t)(m - NCH) * TG + col] += v;
      }
    }
  }
}

// end conv: M = 2*n_half <= 8, K = 256 -> VALU, one thread per (b,t)
__global__ __launch_bounds__(256) void end_conv_kernel(const float* __restrict__ wout,
                                                       const float* __restrict__ ew,
                                                       const float* __restrict__ eb,
                                                       float* __restrict__ obuf,
                                                       int n_half) {
  int tid = blockIdx.x * blockDim.x + threadIdx.x;
  if (tid >= BATCH * TG) return;
  int t = tid & (TG - 1);
  int b = tid >> 11;
  int no = 2 * n_half;
  float o[8];
  for (int j = 0; j < no; ++j) o[j] = eb[j];
  for (int cch = 0; cch < NCH; ++cch) {
    float wv = wout[((size_t)b * NCH + cch) * TG + t];
    for (int j = 0; j < no; ++j) o[j] += ew[j * NCH + cch] * wv;
  }
  for (int j = 0; j < no; ++j) obuf[((size_t)b * 8 + j) * TG + t] = o[j];
}

__global__ __launch_bounds__(256) void coupling_kernel(float* __restrict__ audio,
                                                       const float* __restrict__ obuf,
                                                       float* __restrict__ acc,
                                                       int n_half, int n_rem) {
  __shared__ float red[256];
  int total = BATCH * n_half * TG;
  float lsum = 0.f;
  for (int idx = blockIdx.x * blockDim.x + threadIdx.x; idx < total;
       idx += gridDim.x * blockDim.x) {
    int t = idx & (TG - 1);
    int j = (idx >> 11) % n_half;
    int b = idx / (n_half * TG);
    float s = obuf[((size_t)b * 8 + n_half + j) * TG + t];
    float bi = obuf[((size_t)b * 8 + j) * TG + t];
    size_t ai = ((size_t)b * n_rem + n_half + j) * TG + t;
    audio[ai] = expf(s) * audio[ai] + bi;
    lsum += s;
  }
  red[threadIdx.x] = lsum;
  __syncthreads();
  for (int s = 128; s > 0; s >>= 1) {
    if (threadIdx.x < (unsigned)s) red[threadIdx.x] += red[threadIdx.x + s];
    __syncthreads();
  }
  if (threadIdx.x == 0) atomicAdd(&acc[0], red[0]);
}

__global__ __launch_bounds__(256) void zz_reduce_kernel(const float* __restrict__ z,
                                                        float* __restrict__ acc) {
  __shared__ float red[256];
  float lsum = 0.f;
  for (int i = blockIdx.x * blockDim.x + threadIdx.x; i < BATCH * 8 * TG;
       i += gridDim.x * blockDim.x) {
    float v = z[i];
    lsum += v * v;
  }
  red[threadIdx.x] = lsum;
  __syncthreads();
  for (int s = 128; s > 0; s >>= 1) {
    if (threadIdx.x < (unsigned)s) red[threadIdx.x] += red[threadIdx.x + s];
    __syncthreads();
  }
  if (threadIdx.x == 0) atomicAdd(&acc[2], red[0]);
}

__global__ void finalize_kernel(const float* __restrict__ acc, float* __restrict__ out) {
  if (threadIdx.x == 0 && blockIdx.x == 0)
    out[BATCH * 8 * TG] =
        (acc[2] * 0.5f - acc[0] - acc[1]) * (1.0f / (float)(BATCH * 8 * TG));
}

// ---------------------------------------------------------------------------
extern "C" void kernel_launch(void* const* d_in, const int* in_sizes, int n_in,
                              void* d_out, int out_size, void* d_ws, size_t ws_size,
                              hipStream_t stream) {
  (void)in_sizes; (void)out_size;
  if (n_in < 4 + 12 * 39) return;
  const float* x    = (const float*)d_in[0];
  const float* y    = (const float*)d_in[1];
  const float* up_w = (const float*)d_in[2];
  const float* up_b = (const float*)d_in[3];
  auto FP = [&](int flow, int item) -> const float* {
    return (const float*)d_in[4 + flow * 39 + item];
  };
  // per-flow items: 0=W 1=start_w 2=start_b 3=cond_w 4=cond_b 5=end_w 6=end_b
  // 7+i = in_w[i], 15+i = in_b[i], 23+i = rs_w[i], 31+i = rs_b[i]

  char* base = (char*)d_ws;
  size_t off = 0;
  auto alloc = [&](size_t bytes) -> void* {
    off = (off + 255) & ~(size_t)255;
    void* p = base + off;
    off += bytes;
    return p;
  };
  _Float16* spect16 = (_Float16*)alloc((size_t)BATCH * CONDK * TG * 2);
  float*    audioA  = (float*)alloc((size_t)BATCH * 8 * TG * 4);
  float*    audioB  = (float*)alloc((size_t)BATCH * 8 * TG * 4);
  float*    h32     = (float*)alloc((size_t)BATCH * NCH * TG * 4);
  _Float16* h16     = (_Float16*)alloc((size_t)BATCH * NCH * HPITCH * 2);
  float*    wout    = (float*)alloc((size_t)BATCH * NCH * TG * 4);
  float*    apre    = (float*)alloc((size_t)BATCH * 512 * TG * 4);  // also obuf
  _Float16* acts16  = (_Float16*)alloc((size_t)BATCH * NCH * TG * 2);
  _Float16* Apack   = (_Float16*)alloc((size_t)32 * NKK_LAYER * 32 * 16 * 2);
  _Float16* Rpack   = (_Float16*)alloc((size_t)32 * NKK_RS * 32 * 16 * 2);
  float*    biasb   = (float*)alloc(512 * 4);
  float*    acc     = (float*)alloc(16 * 4);
  if (off > ws_size) return;

  float* zout = (float*)d_out;

  zero_kernel<<<1, 256, 0, stream>>>(acc, 16);
  {
    int n = BATCH * NMEL * 16384;
    upsample_kernel<<<(n + 255) / 256, 256, 0, stream>>>(x, up_w, up_b, spect16);
  }
  {
    int n = BATCH * 8 * TG;
    audio_init_kernel<<<(n + 255) / 256, 256, 0, stream>>>(y, audioA);
  }

  float* cur = audioA;
  float* alt = audioB;
  int n_half = 4, n_rem = 8, zbase = 0;
  const float kLogDetScale = (float)BATCH * (float)TG;

  for (int k = 0; k < 12; ++k) {
    if ((k % 4) == 0 && k > 0) {
      int n_old = n_rem;
      n_half -= 1;
      n_rem -= 2;
      int n = BATCH * n_old * TG;
      early_split_kernel<<<(n + 255) / 256, 256, 0, stream>>>(cur, alt, zout, zbase, n_old);
      zbase += 2;
      float* tmp = cur; cur = alt; alt = tmp;
    }
    logdet_kernel<<<1, 1, 0, stream>>>(FP(k, 0), acc, n_rem, kLogDetScale);
    {
      int n = BATCH * TG;
      wmult_kernel<<<(n + 255) / 256, 256, 0, stream>>>(cur, alt, FP(k, 0), n_rem);
      float* tmp = cur; cur = alt; alt = tmp;
    }
    {
      int nh = BATCH * NCH * HPITCH;
      zero_h16_kernel<<<(nh + 255) / 256, 256, 0, stream>>>(h16, nh);
      int n = BATCH * NCH * TG;
      start_conv_kernel<<<(n + 255) / 256, 256, 0, stream>>>(cur, FP(k, 1), FP(k, 2),
                                                             h32, h16, n_half, n_rem);
      zero_kernel<<<(n + 255) / 256, 256, 0, stream>>>(wout, n);
    }
    for (int i = 0; i < 8; ++i) {
      {
        int n = 32 * NKK_LAYER * 32 * 16;
        pack_layerA_kernel<<<(n + 255) / 256, 256, 0, stream>>>(FP(k, 7 + i), FP(k, 3),
                                                                i, Apack);
      }
      layer_bias_kernel<<<2, 256, 0, stream>>>(FP(k, 15 + i), FP(k, 4), i, biasb);
      layer_gemm_kernel<<<dim3(4, TG / 32, BATCH), 256, 0, stream>>>(
          Apack, h16, spect16, biasb, apre, 1 << i);
      {
        int n = BATCH * NCH * TG;
        gate_kernel<<<(n + 255) / 256, 256, 0, stream>>>(apre, acts16);
      }
      int Mo = (i < 7) ? 512 : 256;
      {
        int n = (Mo / 16) * NKK_RS * 32 * 16;
        pack_rs_kernel<<<(n + 255) / 256, 256, 0, stream>>>(FP(k, 23 + i), Rpack, Mo);
      }
      rs_gemm_kernel<<<dim3(Mo / 128, TG / 32, BATCH), 256, 0, stream>>>(
          Rpack, acts16, FP(k, 31 + i), h32, h16, wout, (i == 7) ? 1 : 0);
    }
    {
      int n = BATCH * TG;
      end_conv_kernel<<<(n + 255) / 256, 256, 0, stream>>>(wout, FP(k, 5), FP(k, 6),
                                                           apre, n_half);
    }
    coupling_kernel<<<64, 256, 0, stream>>>(cur, apre, acc, n_half, n_rem);
  }
  {
    int n = BATCH * 4 * TG;
    final_copy_kernel<<<(n + 255) / 256, 256, 0, stream>>>(cur, zout);
  }
  zz_reduce_kernel<<<128, 256, 0, stream>>>(zout, acc);
  finalize_kernel<<<1, 1, 0, stream>>>(acc, zout);
}
